// FusedQuantizedLinear_34746285425378
// MI455X (gfx1250) — matrix operand order repaired and checked by reading, not compiled, not run on hardware
//
#include <hip/hip_runtime.h>

typedef __attribute__((ext_vector_type(16))) _Float16     v16h;
typedef __attribute__((ext_vector_type(8)))  _Float16     v8h;
typedef __attribute__((ext_vector_type(8)))  float        v8f;
typedef __attribute__((ext_vector_type(4)))  int          v4i;
typedef __attribute__((ext_vector_type(8)))  int          v8i;
typedef __attribute__((ext_vector_type(4)))  unsigned int v4u;

#define BM 256
#define BN 256
#define BK 64             // half a quant group; scale group = kb>>1
#define PITCH 72          // LDS row pitch in halves: 144B rows -> 4*9 bank rotation, b128-conflict-free
#define NTHREADS 512      // 16 wave32s: 4x4 wave grid, 64x64 per wave

__global__ __launch_bounds__(NTHREADS)
void fused_q4_gemm(const _Float16* __restrict__ x,      // [M,K] fp16
                   const int*      __restrict__ weight, // [N,K/2] one packed byte per int32
                   const _Float16* __restrict__ scale,  // [N,K/128] fp16
                   const _Float16* __restrict__ bias,   // [N] fp16
                   _Float16*       __restrict__ out,    // [M,N] fp16
                   int M, int N, int K)
{
    __shared__ _Float16 sA[2][BM * PITCH];   // x tile (TDM-filled, double buffered)
    __shared__ _Float16 sB[BN * PITCH];      // dequantized W tile

    const int tid    = threadIdx.x;
    const int lane   = tid & 31;
    const int wave   = tid >> 5;
    const int wave_m = wave & 3;     // 4 waves along M (64 rows each)
    const int wave_n = wave >> 2;    // 4 waves along N (64 cols each)

    const int m0 = blockIdx.y * BM;
    const int n0 = blockIdx.x * BN;
    const int numGroups = K / 128;
    const int Kh  = K / 2;
    const int nKB = K / BK;

    // ---- TDM: DMA a BM x BK fp16 tile of x into sA[buf] with 72-half row pitch ----
    auto issue_tdm = [&](int kb, int buf) {
        unsigned lds = (unsigned)(size_t)&sA[buf][0];
        unsigned long long ga =
            (unsigned long long)(size_t)(x + (size_t)m0 * K + (size_t)kb * BK);
        // D# group 0: count=1 | lds_addr | global_addr(57b) | type=2
        v4u g0 = { 1u,
                   lds,
                   (unsigned)(ga & 0xffffffffull),
                   (unsigned)((ga >> 32) & 0x01ffffffull) | (2u << 30) };
        // D# group 1: data_size=2B, pad_enable, pad_interval=32 DW (code 4),
        //             pad_amount=4 DW (code 3); dims/strides in elements
        v8i g1 = { (int)((1u << 16) | (1u << 20) | (4u << 22) | (3u << 25)),
                   (int)(((unsigned)K & 0xffffu) << 16),              // tensor_dim0 lo16
                   (int)((((unsigned)K >> 16) & 0xffffu)              // tensor_dim0 hi16
                         | (((unsigned)M & 0xffffu) << 16)),          // tensor_dim1 lo16
                   (int)((((unsigned)M >> 16) & 0xffffu)              // tensor_dim1 hi16
                         | ((unsigned)BK << 16)),                     // tile_dim0 = 64
                   (int)BM,                                           // tile_dim1 = 256
                   (int)K,                                            // tensor_dim0_stride lo32
                   0, 0 };
        v4i gz = { 0, 0, 0, 0 };
#if __clang_major__ >= 23
        v8i gz8 = { 0, 0, 0, 0, 0, 0, 0, 0 };
        __builtin_amdgcn_tensor_load_to_lds(g0, g1, gz, gz, gz8, 0);
#else
        __builtin_amdgcn_tensor_load_to_lds(g0, g1, gz, gz, 0);
#endif
    };

    // ---- packed-weight staging decomposition: 2 threads per n-row ----
    const int wn   = tid >> 1;       // 0..255
    const int wsel = tid & 1;        // 32-half (16 int32) half of the k-block

    auto load_w = [&](int kb, v4i* wreg, float& s) {
        const v4i* wsrc = (const v4i*)(weight + (size_t)(n0 + wn) * Kh
                                       + kb * (BK / 2) + wsel * 16);
#pragma unroll
        for (int g = 0; g < 4; ++g) wreg[g] = wsrc[g];
        s = (float)scale[(size_t)(n0 + wn) * numGroups + (kb >> 1)];
    };

    auto store_w = [&](const v4i* wreg, float s) {
        _Float16* dst = &sB[wn * PITCH + wsel * 32];
#pragma unroll
        for (int g = 0; g < 4; ++g) {
            v8h o;
#pragma unroll
            for (int j = 0; j < 4; ++j) {
                int w = wreg[g][j];
                o[2 * j]     = (_Float16)((float)((w & 15) - 8) * s);
                o[2 * j + 1] = (_Float16)((float)(((w >> 4) & 15) - 8) * s);
            }
            *(v8h*)(dst + g * 8) = o;
        }
    };

    const v8f vzero = {0.f, 0.f, 0.f, 0.f, 0.f, 0.f, 0.f, 0.f};
    v8f acc[4][4];
#pragma unroll
    for (int mt = 0; mt < 4; ++mt)
#pragma unroll
        for (int nt = 0; nt < 4; ++nt) acc[mt][nt] = vzero;

    // ---------------- prologue: stage block 0 ----------------
    if (wave == 0) issue_tdm(0, 0);
    {
        v4i wreg[4];
        float s;
        load_w(0, wreg, s);
        store_w(wreg, s);
    }
    if (wave == 0) __builtin_amdgcn_s_wait_tensorcnt(0);
    __syncthreads();

    const int frow   = lane & 15;
    const int asel8  = (lane >> 4) * 8;    // A frag: hi half-wave at k+8 / k+24
    const int bsel16 = (lane >> 4) * 8;    // B frag: hi half-wave at k+8 / k+24

    // ---------------- main pipelined loop over K-blocks ----------------
    for (int kb = 0; kb < nKB; ++kb) {
        const int  cur  = kb & 1;
        const int  nxt  = cur ^ 1;
        const bool more = (kb + 1 < nKB);

        if (more && wave == 0) issue_tdm(kb + 1, nxt);   // DMA next A tile (async)
        v4i   wreg[4];
        float s = 0.f;
        if (more) load_w(kb + 1, wreg, s);               // next W packed -> regs (in flight)

        // compute current block: 2 k-steps of 32, 16 WMMAs each
        const _Float16* a = &sA[cur][0];
#pragma unroll
        for (int ks = 0; ks < 2; ++ks) {
            const int kk = ks * 32;

            v16h afrag[4];
#pragma unroll
            for (int mt = 0; mt < 4; ++mt) {
                int r = wave_m * 64 + mt * 16 + frow;
                v8h lo = *(const v8h*)&a[r * PITCH + kk + asel8];
                v8h hi = *(const v8h*)&a[r * PITCH + kk + asel8 + 16];
                afrag[mt] = __builtin_shufflevector(lo, hi,
                    0, 1, 2, 3, 4, 5, 6, 7, 8, 9, 10, 11, 12, 13, 14, 15);
            }

#pragma unroll
            for (int nt = 0; nt < 4; ++nt) {
                int n = wave_n * 64 + nt * 16 + frow;
                v8h lo = *(const v8h*)&sB[n * PITCH + kk + bsel16];
                v8h hi = *(const v8h*)&sB[n * PITCH + kk + bsel16 + 16];
                v16h bfrag = __builtin_shufflevector(lo, hi,
                    0, 1, 2, 3, 4, 5, 6, 7, 8, 9, 10, 11, 12, 13, 14, 15);
#pragma unroll
                for (int mt = 0; mt < 4; ++mt)
                    acc[mt][nt] = __builtin_amdgcn_wmma_f32_16x16x32_f16(
                        false, afrag[mt], false, bfrag, (short)0, acc[mt][nt],
                        false, false);
            }
        }

        __syncthreads();                         // all waves done reading sB / sA[cur]
        if (more) store_w(wreg, s);              // dequant next W -> sB
        if (more && wave == 0) __builtin_amdgcn_s_wait_tensorcnt(0);
        __syncthreads();                         // next A tile + next sB visible
    }

    // ---------------- epilogue: + bias, non-temporal fp16 stores ----------------
    // Loop order mt -> r -> nt: the 4 nt stores per row hit one 128B line
    // back-to-back (lanes 0-15) so L2 merges full lines; NT hint keeps the
    // streamed output from evicting the L2-resident x/weight working set.
    const int ccol  = lane & 15;
    const int rbase = (lane >> 4) * 8;
    float bv[4];
#pragma unroll
    for (int nt = 0; nt < 4; ++nt)
        bv[nt] = (float)bias[n0 + wave_n * 64 + nt * 16 + ccol];

    _Float16* obase = out + (size_t)(m0 + wave_m * 64 + rbase) * N
                          + n0 + wave_n * 64 + ccol;
#pragma unroll
    for (int mt = 0; mt < 4; ++mt) {
#pragma unroll
        for (int r = 0; r < 8; ++r) {
            _Float16* orow = obase + (size_t)(mt * 16 + r) * N;
#pragma unroll
            for (int nt = 0; nt < 4; ++nt)
                __builtin_nontemporal_store(
                    (_Float16)(acc[mt][nt][r] + bv[nt]), orow + nt * 16);
        }
    }
}

extern "C" void kernel_launch(void* const* d_in, const int* in_sizes, int n_in,
                              void* d_out, int out_size, void* d_ws, size_t ws_size,
                              hipStream_t stream)
{
    const _Float16* x      = (const _Float16*)d_in[0];
    const int*      weight = (const int*)d_in[1];
    const _Float16* scale  = (const _Float16*)d_in[2];
    const _Float16* bias   = (const _Float16*)d_in[3];
    _Float16*       out    = (_Float16*)d_out;

    const int N = in_sizes[3];                                   // 11008
    const int K = (int)((2LL * (long long)in_sizes[1]) / N);     // 4096
    const int M = in_sizes[0] / K;                               // 8192

    dim3 grid(N / BN, M / BM);   // 43 x 32
    fused_q4_gemm<<<grid, NTHREADS, 0, stream>>>(x, weight, scale, bias, out, M, N, K);
}
